// GeneralAttention_52381421142030
// MI455X (gfx1250) — matrix operand order repaired and checked by reading, not compiled, not run on hardware
//
#include <hip/hip_runtime.h>
#include <hip/hip_bf16.h>

// Problem constants (match reference)
constexpr int B = 16;
constexpr int L = 1024;
constexpr int D = 512;
constexpr float SCALE = 0.04419417382415922f; // 1/sqrt(512)

typedef __bf16 bf16_t;
typedef __attribute__((ext_vector_type(16))) __bf16 v16bf;
typedef __attribute__((ext_vector_type(8)))  __bf16 v8bf;
typedef __attribute__((ext_vector_type(8)))  float  v8f;
typedef __attribute__((ext_vector_type(4)))  float  v4f;

// Padded LDS row stride for the staged B panel: 40 elems = 80 B.
//  - keeps every per-lane 32 B column read 16 B aligned
//  - lane n's bank offset = n*20 mod 64 -> all multiples of 4, minimal conflict
constexpr int BSTRIDE = 40;
constexpr int LDS_B_ELEMS = 64 * BSTRIDE;   // 5120 B per buffer (x2: double buffer)

// ---------------------------------------------------------------------------
// WMMA tile loaders (gfx1250 wave32 layouts, CDNA5 ISA §7.12.2)
//
// A (16x32 bf16, MxK): lane = hl*16 + m ; holds row m:
//   elems 0..7  -> K = k0 + hl*8 + 0..7
//   elems 8..15 -> K = k0 + 16 + hl*8 + 0..7
// B (32x16 bf16, KxN): lane = hl*16 + n ; holds col n:
//   elems 0..7 -> K = k0 + hl*8 + 0..7 ; elems 8..15 -> K = k0 + 16 + hl*8 + 0..7 (as A)
// C/D (16x16 f32): lane = hl*16 + n ; vgpr r -> row = hl*8 + r, col = n
// ---------------------------------------------------------------------------

__device__ __forceinline__ v16bf load_a_bf16(const bf16_t* base, int lda, int k0, int lane) {
    int hl = lane >> 4, m = lane & 15;
    const bf16_t* p = base + (size_t)m * lda + k0;
    v8bf lo = *(const v8bf*)(p + hl * 8);
    v8bf hi = *(const v8bf*)(p + 16 + hl * 8);
    v16bf a;
#pragma unroll
    for (int i = 0; i < 8; ++i) { a[i] = lo[i]; a[i + 8] = hi[i]; }
    return a;
}

// A-tile from an f32 source, converted to bf16 in registers (for attn @ inp)
__device__ __forceinline__ v16bf load_a_f32_cvt(const float* base, int lda, int k0, int lane) {
    int hl = lane >> 4, m = lane & 15;
    const float* p = base + (size_t)m * lda + k0;
    v4f x0 = *(const v4f*)(p + hl * 8);
    v4f x1 = *(const v4f*)(p + hl * 8 + 4);
    v4f x2 = *(const v4f*)(p + 16 + hl * 8);
    v4f x3 = *(const v4f*)(p + 16 + hl * 8 + 4);
    v16bf a;
#pragma unroll
    for (int i = 0; i < 4; ++i) {
        a[i]      = (bf16_t)x0[i];
        a[i + 4]  = (bf16_t)x1[i];
        a[i + 8]  = (bf16_t)x2[i];
        a[i + 12] = (bf16_t)x3[i];
    }
    return a;
}

// ---------------------------------------------------------------------------
// Shared GEMM core: 4-wave block computes a 64x64 tile of C.
// Wave w owns rows [w*16, w*16+16); all waves share a 64-row B panel staged
// in a double-buffered LDS region, processed as explicit phase pairs
// (buf0->buf1, buf1->buf0) so LDS offsets are compile-time constants.
// Each phase: issue A loads, issue next-panel global loads (held in regs),
// run ds_loads + 4 WMMAs on the current panel, then commit the register-held
// panel to the other buffer and barrier — prefetch latency hides behind math.
// Aptr must already point at this wave's 16-row A strip.
// ---------------------------------------------------------------------------
template<bool A_F32>
__device__ __forceinline__ void gemm_core(const void* Aptr, int lda,
                                          const bf16_t* Bbase, int ldb, int n0,
                                          int kdim, bf16_t* ldsB, v8f* acc) {
    const int tid  = threadIdx.x;      // 0..127
    const int lane = tid & 31;
    const int hl = lane >> 4, n = lane & 15;
    // This thread's two 16 B fill chunks (64 rows x 4 segs = 256 chunks).
    const int row0 = (tid * 2) >> 2,     seg0 = (tid * 2) & 3;
    const int row1 = (tid * 2 + 1) >> 2, seg1 = (tid * 2 + 1) & 3;
    const bf16_t* gsrc0 = Bbase + (size_t)(n0 + row0) * ldb + seg0 * 8;
    const bf16_t* gsrc1 = Bbase + (size_t)(n0 + row1) * ldb + seg1 * 8;
    const int l0off = row0 * BSTRIDE + seg0 * 8;
    const int l1off = row1 * BSTRIDE + seg1 * 8;

    // Prologue: fill buffer 0 with panel k=0.
    {
        v8bf c0 = *(const v8bf*)(gsrc0);
        v8bf c1 = *(const v8bf*)(gsrc1);
        *(v8bf*)(ldsB + l0off) = c0;
        *(v8bf*)(ldsB + l1off) = c1;
    }
    __syncthreads();

    auto phase = [&](int kc, const bf16_t* cur, bf16_t* nxt, bool pf) {
        // A tile FIRST: LOADcnt decrements in order, so waiting on A must not
        // imply waiting on the prefetch loads issued after it.
        v16bf a;
        if constexpr (A_F32) a = load_a_f32_cvt((const float*)Aptr, lda, kc, lane);
        else                 a = load_a_bf16((const bf16_t*)Aptr, lda, kc, lane);
        // Next-panel global loads: held in registers until after the WMMAs.
        v8bf c0 = {}, c1 = {};
        if (pf) {
            c0 = *(const v8bf*)(gsrc0 + kc + 32);
            c1 = *(const v8bf*)(gsrc1 + kc + 32);
        }
        // Batch all four B-tiles into registers, then WMMAs back-to-back.
        v16bf bt[4];
#pragma unroll
        for (int t = 0; t < 4; ++t) {
            const bf16_t* p = cur + (t * 16 + n) * BSTRIDE + hl * 8;
            v8bf lo = *(const v8bf*)(p);
            v8bf hi = *(const v8bf*)(p + 16);
#pragma unroll
            for (int i = 0; i < 8; ++i) { bt[t][i] = lo[i]; bt[t][i + 8] = hi[i]; }
        }
#pragma unroll
        for (int t = 0; t < 4; ++t)
            acc[t] = __builtin_amdgcn_wmma_f32_16x16x32_bf16(
                false, a, false, bt[t], (short)0, acc[t], false, false);
        // Commit prefetched panel late: global-load wait lands here, after
        // the math, instead of stalling before it.
        if (pf) {
            *(v8bf*)(nxt + l0off) = c0;
            *(v8bf*)(nxt + l1off) = c1;
        }
        __syncthreads();
    };

    bf16_t* buf0 = ldsB;
    bf16_t* buf1 = ldsB + LDS_B_ELEMS;
    for (int k0 = 0; k0 < kdim; k0 += 64) {        // kdim is a multiple of 64
        phase(k0,      buf0, buf1, true);
        phase(k0 + 32, buf1, buf0, k0 + 64 < kdim);
    }
}

// ---------------------------------------------------------------------------
// Pass 0a: f32 -> bf16 convert of inp, plus LDS-tiled transpose (inpT[b][d][l])
// so the final GEMM's B-panels are contiguous loads. Both streams coalesced.
// ---------------------------------------------------------------------------
__global__ void __launch_bounds__(256)
convert_transpose_kernel(const float* __restrict__ inp,
                         bf16_t* __restrict__ inp_bf,
                         bf16_t* __restrict__ inpT_bf) {
    __shared__ bf16_t tile[32][33];
    int b  = blockIdx.z;
    int l0 = blockIdx.x * 32;
    int d0 = blockIdx.y * 32;
    int tx = threadIdx.x;           // 0..31
    int ty = threadIdx.y;           // 0..7
    const float* src = inp + (size_t)b * L * D;
    bf16_t* dst   = inp_bf  + (size_t)b * L * D;
    bf16_t* dstT  = inpT_bf + (size_t)b * D * L;
#pragma unroll
    for (int kk = 0; kk < 4; ++kk) {
        int l = l0 + ty + kk * 8;
        int d = d0 + tx;
        bf16_t v = (bf16_t)src[(size_t)l * D + d];
        dst[(size_t)l * D + d] = v;
        tile[ty + kk * 8][tx] = v;
    }
    __syncthreads();
#pragma unroll
    for (int kk = 0; kk < 4; ++kk) {
        int d = d0 + ty + kk * 8;
        int l = l0 + tx;
        dstT[(size_t)d * L + l] = tile[tx][ty + kk * 8];
    }
}

// Pass 0b: W f32 -> bf16 (row-major; W rows are the K-runs for the first GEMM)
__global__ void __launch_bounds__(256)
convert_w_kernel(const float* __restrict__ W, bf16_t* __restrict__ W_bf) {
    int i = blockIdx.x * 256 + threadIdx.x;
    W_bf[i] = (bf16_t)W[i];
}

// ---------------------------------------------------------------------------
// Pass 1: context[b] = inp[b] @ W^T   (bf16 in, bf16 out, f32 accumulate)
// ---------------------------------------------------------------------------
__global__ void __launch_bounds__(128)
ctx_gemm_kernel(const bf16_t* __restrict__ inp_bf,
                const bf16_t* __restrict__ W_bf,
                bf16_t* __restrict__ ctx_bf) {
    __shared__ bf16_t ldsB[2 * LDS_B_ELEMS];
    int tid = threadIdx.x, lane = tid & 31, w = tid >> 5;
    int b  = blockIdx.z;
    int m0 = blockIdx.x * 64 + w * 16;
    int n0 = blockIdx.y * 64;
    const bf16_t* A = inp_bf + (size_t)b * L * D + (size_t)m0 * D;
    v8f acc[4] = {};
    gemm_core<false>(A, D, W_bf, D, n0, D, ldsB, acc);
    int hl = lane >> 4, col = lane & 15;
    bf16_t* C = ctx_bf + (size_t)b * L * D;
#pragma unroll
    for (int t = 0; t < 4; ++t)
#pragma unroll
        for (int r = 0; r < 8; ++r) {
            int row = m0 + hl * 8 + r;
            C[(size_t)row * D + n0 + t * 16 + col] = (bf16_t)acc[t][r];
        }
}

// ---------------------------------------------------------------------------
// Pass 2: logits[b] = (inp[b] @ context[b]^T) * (1/sqrt(D))  -> f32 into d_out
// ---------------------------------------------------------------------------
__global__ void __launch_bounds__(128)
logits_gemm_kernel(const bf16_t* __restrict__ inp_bf,
                   const bf16_t* __restrict__ ctx_bf,
                   float* __restrict__ attn) {
    __shared__ bf16_t ldsB[2 * LDS_B_ELEMS];
    int tid = threadIdx.x, lane = tid & 31, w = tid >> 5;
    int b  = blockIdx.z;
    int m0 = blockIdx.x * 64 + w * 16;   // q
    int n0 = blockIdx.y * 64;            // k
    const bf16_t* A  = inp_bf + (size_t)b * L * D + (size_t)m0 * D;
    const bf16_t* Bm = ctx_bf + (size_t)b * L * D;
    v8f acc[4] = {};
    gemm_core<false>(A, D, Bm, D, n0, D, ldsB, acc);
    int hl = lane >> 4, col = lane & 15;
    float* C = attn + (size_t)b * L * L;
#pragma unroll
    for (int t = 0; t < 4; ++t)
#pragma unroll
        for (int r = 0; r < 8; ++r) {
            int row = m0 + hl * 8 + r;
            C[(size_t)row * L + n0 + t * 16 + col] = acc[t][r] * SCALE;
        }
}

// ---------------------------------------------------------------------------
// Pass 3: masked row softmax over attn (in place, f32). One 256-thread block
// (8 wave32s) per (b,q) row of length L=1024; shuffle + LDS reductions.
// ---------------------------------------------------------------------------
__global__ void __launch_bounds__(256)
softmax_kernel(float* __restrict__ attn, const unsigned char* __restrict__ mask) {
    __shared__ float smax[8];
    __shared__ float ssum[8];
    size_t row = blockIdx.x;                 // b*L + q
    float* p = attn + row * L;
    const unsigned char* mrow = mask + row * L;
    int tid = threadIdx.x;
    int wid = tid >> 5;

    float vals[4];
    float mx = -INFINITY;
#pragma unroll
    for (int i = 0; i < 4; ++i) {
        int k = tid + i * 256;
        float v = p[k];
        if (mrow[k]) v = -INFINITY;
        vals[i] = v;
        mx = fmaxf(mx, v);
    }
#pragma unroll
    for (int off = 16; off > 0; off >>= 1)
        mx = fmaxf(mx, __shfl_xor(mx, off, 32));
    if ((tid & 31) == 0) smax[wid] = mx;
    __syncthreads();
    float rowmax = smax[0];
#pragma unroll
    for (int j = 1; j < 8; ++j) rowmax = fmaxf(rowmax, smax[j]);

    float s = 0.0f;
#pragma unroll
    for (int i = 0; i < 4; ++i) {
        vals[i] = expf(vals[i] - rowmax);
        s += vals[i];
    }
#pragma unroll
    for (int off = 16; off > 0; off >>= 1)
        s += __shfl_xor(s, off, 32);
    if ((tid & 31) == 0) ssum[wid] = s;
    __syncthreads();
    float rowsum = ssum[0];
#pragma unroll
    for (int j = 1; j < 8; ++j) rowsum += ssum[j];
    float inv = 1.0f / rowsum;
#pragma unroll
    for (int i = 0; i < 4; ++i)
        p[tid + i * 256] = vals[i] * inv;
}

// ---------------------------------------------------------------------------
// Pass 4: out[b] = attn[b] @ inp[b]. A from f32 attn (cvt to bf16 in regs),
// B from transposed bf16 inp (inpT[d][k] contiguous along k).
// ---------------------------------------------------------------------------
__global__ void __launch_bounds__(128)
out_gemm_kernel(const float* __restrict__ attn,
                const bf16_t* __restrict__ inpT_bf,
                float* __restrict__ out) {
    __shared__ bf16_t ldsB[2 * LDS_B_ELEMS];
    int tid = threadIdx.x, lane = tid & 31, w = tid >> 5;
    int b  = blockIdx.z;
    int m0 = blockIdx.x * 64 + w * 16;   // q
    int n0 = blockIdx.y * 64;            // d
    const float*  A  = attn    + (size_t)b * L * L + (size_t)m0 * L;
    const bf16_t* Bm = inpT_bf + (size_t)b * D * L;
    v8f acc[4] = {};
    gemm_core<true>(A, L, Bm, L, n0, L, ldsB, acc);
    int hl = lane >> 4, col = lane & 15;
    float* C = out + (size_t)b * L * D;
#pragma unroll
    for (int t = 0; t < 4; ++t)
#pragma unroll
        for (int r = 0; r < 8; ++r) {
            int row = m0 + hl * 8 + r;
            C[(size_t)row * D + n0 + t * 16 + col] = acc[t][r];
        }
}

// ---------------------------------------------------------------------------
extern "C" void kernel_launch(void* const* d_in, const int* in_sizes, int n_in,
                              void* d_out, int out_size, void* d_ws, size_t ws_size,
                              hipStream_t stream) {
    const float* inp          = (const float*)d_in[0];          // [B,L,D] f32
    const unsigned char* mask = (const unsigned char*)d_in[1];  // [B,L,L] bool
    const float* W            = (const float*)d_in[2];          // [D,D]   f32

    float* out  = (float*)d_out;                        // [B,L,D] f32
    float* attn = (float*)d_out + (size_t)B * L * D;    // [B,L,L] f32

    // Workspace layout (bf16): inp_bf | inpT_bf | ctx_bf | W_bf  (~51 MB)
    bf16_t* inp_bf  = (bf16_t*)d_ws;
    bf16_t* inpT_bf = inp_bf  + (size_t)B * L * D;
    bf16_t* ctx_bf  = inpT_bf + (size_t)B * L * D;
    bf16_t* W_bf    = ctx_bf  + (size_t)B * L * D;

    // Pass 0: convert + transpose
    convert_transpose_kernel<<<dim3(L / 32, D / 32, B), dim3(32, 8), 0, stream>>>(
        inp, inp_bf, inpT_bf);
    convert_w_kernel<<<(D * D) / 256, 256, 0, stream>>>(W, W_bf);

    // Pass 1: context = inp @ W^T (bf16)
    ctx_gemm_kernel<<<dim3(L / 64, D / 64, B), 128, 0, stream>>>(inp_bf, W_bf, ctx_bf);

    // Pass 2: scaled logits -> attn region of d_out (f32)
    logits_gemm_kernel<<<dim3(L / 64, L / 64, B), 128, 0, stream>>>(inp_bf, ctx_bf, attn);

    // Pass 3: masked softmax in place
    softmax_kernel<<<B * L, 256, 0, stream>>>(attn, mask);

    // Pass 4: out = attn @ inp
    out_gemm_kernel<<<dim3(L / 64, D / 64, B), 128, 0, stream>>>(attn, inpT_bf, out);
}
